// DenseKAN_81655918231937
// MI455X (gfx1250) — matrix operand order repaired and checked by reading, not compiled, not run on hardware
//
#include <hip/hip_runtime.h>

typedef __attribute__((ext_vector_type(16))) _Float16 v16h;
typedef __attribute__((ext_vector_type(8)))  float    v8f;

#define UNITS          128
#define INPUT_DIM      128
#define BATCH          4096
#define ROWS_PER_BLOCK 32
#define LDS_STRIDE     264   // halfs per row (528 B = 132 dwords; 132 % 64 = 4 banks/row)

// ---------------------------------------------------------------------------
// Prep: fused weight W[K=2048][N=128] in f16, stored directly in WMMA
// B-fragment order:  wopt[ct(8)][ks(64)][lane(32)][j(16)]
//   lane<16 : N = ct*16+lane,     K = ks*32 + j      (j = 0..15)
//   lane>=16: N = ct*16+lane-16,  K = ks*32 + 16 + j
// Per input dim i (16 K-slots): k<8 -> sk[i][k][o]*scale[i][o],
//                               k==8 -> scale[i][o] (silu residual), else 0.
// ---------------------------------------------------------------------------
__global__ void kan_fuse_weights(const float* __restrict__ sk,     // (128,8,128)
                                 const float* __restrict__ scale,  // (128,128)
                                 _Float16* __restrict__ wopt)      // 262144 halfs
{
    int idx = blockIdx.x * blockDim.x + threadIdx.x;   // 262144 threads
    int j   = idx & 15;
    int L   = (idx >> 4) & 31;
    int ks  = (idx >> 9) & 63;
    int ct  = idx >> 15;
    int K   = ks * 32 + ((L < 16) ? 0 : 16) + j;
    int N   = ct * 16 + (L & 15);
    int i   = K >> 4;
    int k   = K & 15;
    float v = 0.0f;
    if (k < 8)       v = sk[(i * 8 + k) * UNITS + N] * scale[i * UNITS + N];
    else if (k == 8) v = scale[i * UNITS + N];
    wopt[idx] = (_Float16)v;
}

// ---------------------------------------------------------------------------
// Main: per block = 32 batch rows, 8 wave32.
//   wave -> row-tile rt = wave>>2 (16 rows), col-tiles ct0 = wave&3, ct1 = ct0+4
// 8 K-chunks of 256 (16 input dims each): compute bases -> LDS (A-frag layout),
// then 8 WMMA k-steps x 2 col tiles per wave.
// ---------------------------------------------------------------------------
__launch_bounds__(256, 2)
__global__ void kan_wmma(const float* __restrict__ x,      // (4096,128)
                         const float* __restrict__ bias,   // (128)
                         const _Float16* __restrict__ wopt,
                         float* __restrict__ out)          // (4096,128)
{
    __shared__ _Float16 As[ROWS_PER_BLOCK * LDS_STRIDE];

    const int tid     = threadIdx.x;
    const int lane    = tid & 31;
    const int wave    = tid >> 5;                 // 0..7
    const int rowBase = blockIdx.x * ROWS_PER_BLOCK;

    const int rt  = wave >> 2;                    // 0..1
    const int ct0 = wave & 3;                     // 0..3
    const int ct1 = ct0 + 4;                      // 4..7

    v8f c0 = {};
    v8f c1 = {};

    for (int kc = 0; kc < 8; ++kc) {
        // ---- Phase 1: bases + silu for 32 rows x 16 dims (2 elems/thread) ----
        #pragma unroll
        for (int rep = 0; rep < 2; ++rep) {
            int e  = tid + rep * 256;             // 0..511
            int bl = e >> 4;                      // local row 0..31
            int li = e & 15;                      // dim-in-chunk 0..15
            float xv = x[(rowBase + bl) * INPUT_DIM + kc * 16 + li];

            // Cox-de Boor on uniform knots t_j = -2.2 + 0.4*j  (j = 0..11)
            float bb[11];
            #pragma unroll
            for (int j2 = 0; j2 < 11; ++j2) {
                float t0 = -2.2f + 0.4f * (float)j2;
                float t1 = t0 + 0.4f;
                bb[j2] = (xv >= t0 && xv < t1) ? 1.0f : 0.0f;
            }
            #pragma unroll
            for (int k2 = 1; k2 <= 3; ++k2) {
                float inv = 1.0f / (0.4f * (float)k2);
                #pragma unroll
                for (int j2 = 0; j2 + k2 < 11; ++j2) {
                    float tj    = -2.2f + 0.4f * (float)j2;
                    float left  = (xv - tj) * inv;
                    float right = ((tj + 0.4f * (float)(k2 + 1)) - xv) * inv;
                    bb[j2] = left * bb[j2] + right * bb[j2 + 1];
                }
            }
            float s = xv / (1.0f + __expf(-xv));  // silu residual

            _Float16 h[16];
            #pragma unroll
            for (int k2 = 0; k2 < 8; ++k2) h[k2] = (_Float16)bb[k2];
            h[8] = (_Float16)s;
            #pragma unroll
            for (int k2 = 9; k2 < 16; ++k2) h[k2] = (_Float16)0.0f;

            _Float16* dst = &As[bl * LDS_STRIDE + li * 16];
            *(uint4*)(dst)     = *(const uint4*)&h[0];   // ds_store_b128
            *(uint4*)(dst + 8) = *(const uint4*)&h[8];   // ds_store_b128
        }
        __syncthreads();

        // ---- Phase 2: 8 WMMA k-steps over this 256-wide chunk ----
        #pragma unroll
        for (int ks = 0; ks < 8; ++ks) {
            // A fragment (16-bit A 16x32 layout, contiguous K to match B):
            //   lane<16 : M=lane,    VGPR0-7 = K 0..15
            //   lane>=16: M=lane-16, VGPR0-7 = K 16..31
            int arow = rt * 16 + (lane & 15);
            int koff = ks * 32 + ((lane < 16) ? 0 : 16);
            const _Float16* ap = &As[arow * LDS_STRIDE + koff];
            v16h a;
            ((uint4*)&a)[0] = *(const uint4*)(ap);        // ds_load_b128
            ((uint4*)&a)[1] = *(const uint4*)(ap + 8);    // ds_load_b128

            int gks = kc * 8 + ks;                        // global k-step 0..63
            const uint4* bp0 = (const uint4*)(wopt + ((size_t)((ct0 * 64 + gks) * 32 + lane) << 4));
            const uint4* bp1 = (const uint4*)(wopt + ((size_t)((ct1 * 64 + gks) * 32 + lane) << 4));
            v16h b0, b1;
            ((uint4*)&b0)[0] = bp0[0];                    // global_load_b128
            ((uint4*)&b0)[1] = bp0[1];
            ((uint4*)&b1)[0] = bp1[0];
            ((uint4*)&b1)[1] = bp1[1];

            c0 = __builtin_amdgcn_wmma_f32_16x16x32_f16(
                    false, a, false, b0, (short)0, c0, false, false);
            c1 = __builtin_amdgcn_wmma_f32_16x16x32_f16(
                    false, a, false, b1, (short)0, c1, false, false);
        }
        __syncthreads();
    }

    // ---- Epilogue: C layout -> rows, add bias, coalesced stores ----
    int ncol0 = ct0 * 16 + (lane & 15);
    int ncol1 = ct1 * 16 + (lane & 15);
    float bias0 = bias[ncol0];
    float bias1 = bias[ncol1];
    int m0 = (lane < 16) ? 0 : 8;
    #pragma unroll
    for (int r = 0; r < 8; ++r) {
        int row = rowBase + rt * 16 + m0 + r;
        out[row * UNITS + ncol0] = c0[r] + bias0;
        out[row * UNITS + ncol1] = c1[r] + bias1;
    }
}

extern "C" void kernel_launch(void* const* d_in, const int* in_sizes, int n_in,
                              void* d_out, int out_size, void* d_ws, size_t ws_size,
                              hipStream_t stream) {
    (void)in_sizes; (void)n_in; (void)out_size; (void)ws_size;

    const float* x     = (const float*)d_in[0];   // (4096,128)
    const float* sk    = (const float*)d_in[1];   // (128,8,128)
    const float* scale = (const float*)d_in[2];   // (128,128)
    const float* bias  = (const float*)d_in[3];   // (128)
    float*       out   = (float*)d_out;           // (4096,128)
    _Float16*    wopt  = (_Float16*)d_ws;         // 262144 halfs = 512 KB

    // Build fused, fragment-ordered f16 weights (runs every call: deterministic).
    kan_fuse_weights<<<262144 / 256, 256, 0, stream>>>(sk, scale, wopt);

    // Main WMMA GEMM: 4096 rows / 32 per block.
    kan_wmma<<<BATCH / ROWS_PER_BLOCK, 256, 0, stream>>>(x, bias, wopt, out);
}
